// LinearCrossAttention_22316650070090
// MI455X (gfx1250) — matrix operand order repaired and checked by reading, not compiled, not run on hardware
//
#include <hip/hip_runtime.h>
#include <hip/hip_bf16.h>

// ---------------------------------------------------------------------------
// LinearCrossAttention fused kernel for MI455X (gfx1250, wave32, WMMA)
//   B=65536, DIM=512, CTX_DIM=768, HEADS=4, DIM_HEAD=32, HID=128
// ---------------------------------------------------------------------------

typedef __attribute__((ext_vector_type(16))) __bf16 v16bf;
typedef __attribute__((ext_vector_type(8)))  float  v8f;

#define DIM_   512
#define CTX_   768
#define HID_   128
#define MTILE  64          // batch rows per workgroup
#define WAVES  4           // 16 rows per wave
#define QSTR   132         // padded LDS row stride (floats) for 128-wide tiles
#define YSTR   516         // padded LDS row stride (floats) for 512-wide tiles
#define SCALE_ATTN 0.17677669529663687f   // 32^-0.5
#define SQRT_DIM   22.62741699796952f     // 512^0.5

// ---------------------------------------------------------------------------
// Weight repack: W[K][N] f32 row-major  ->  bf16 B-matrix tiles (32x16) in the
// CDNA5 WMMA per-lane layout: within tile (kt,nt):
//   lane 0..15  : N = nt*16 + lane,      e=0..7 -> K = kt*32 + e,     e=8..15 -> K = kt*32 + 16 + (e-8)
//   lane 16..31 : N = nt*16 + (lane-16), e=0..7 -> K = kt*32 + 8 + e, e=8..15 -> K = kt*32 + 24 + (e-8)
// Tiles stored contiguously: [(kt*NT + nt)*32 + lane]*16 + e
// ---------------------------------------------------------------------------
__global__ void pack_b_kernel(const float* __restrict__ W, __bf16* __restrict__ P,
                              int K, int N) {
    int idx = blockIdx.x * blockDim.x + threadIdx.x;
    int total = K * N;
    if (idx >= total) return;
    int elem = idx & 15;
    int lane = (idx >> 4) & 31;
    int t    = idx >> 9;           // tile id = kt*NT + nt
    int NT   = N >> 4;
    int kt   = t / NT;
    int nt   = t - kt * NT;
    int k = kt * 32 + ((lane >> 4) << 3) + (elem & 7) + ((elem >> 3) << 4);
    int n = nt * 16 + (lane & 15);
    P[idx] = (__bf16)W[(size_t)k * N + n];
}

// Build a 16x32 bf16 A tile from 16 row-major f32 values at p (this lane's
// slice): elements 0..7 <- p[0..7], elements 8..15 <- p[16..23].
// Caller guarantees 16B alignment of p.
__device__ __forceinline__ v16bf make_a(const float* p) {
    const float4* q = (const float4*)p;
    float4 c0 = q[0], c1 = q[1], c2 = q[4], c3 = q[5];
    v16bf a;
    a[0]  = (__bf16)c0.x; a[1]  = (__bf16)c0.y; a[2]  = (__bf16)c0.z; a[3]  = (__bf16)c0.w;
    a[4]  = (__bf16)c1.x; a[5]  = (__bf16)c1.y; a[6]  = (__bf16)c1.z; a[7]  = (__bf16)c1.w;
    a[8]  = (__bf16)c2.x; a[9]  = (__bf16)c2.y; a[10] = (__bf16)c2.z; a[11] = (__bf16)c2.w;
    a[12] = (__bf16)c3.x; a[13] = (__bf16)c3.y; a[14] = (__bf16)c3.z; a[15] = (__bf16)c3.w;
    return a;
}

// Load a packed 32x16 bf16 B tile slice for this lane (32 contiguous bytes).
__device__ __forceinline__ v16bf load_b(const __bf16* __restrict__ P, int NT,
                                        int kk, int nt, int lane) {
    return *(const v16bf*)(P + ((size_t)((kk * NT + nt) * 32 + lane) << 4));
}

__device__ __forceinline__ float sigmoid_f(float x) {
    return 1.0f / (1.0f + __expf(-x));
}

__global__ __launch_bounds__(WAVES * 32)
void lca_fused_kernel(const float* __restrict__ x,    // [B,512]
                      const float* __restrict__ ctx,  // [B,768]
                      const __bf16* __restrict__ WqP, // packed 512x128
                      const __bf16* __restrict__ WkP, // packed 768x128
                      const __bf16* __restrict__ WvP, // packed 768x128
                      const __bf16* __restrict__ WoP, // packed 128x512
                      const float* __restrict__ bo,   // [512]
                      const float* __restrict__ gg,   // [512]
                      float* __restrict__ out,        // [B,512]
                      int rows) {
    extern __shared__ float smem[];
    const int wave = (int)(threadIdx.x >> 5);
    const int lane = (int)(threadIdx.x & 31);
    const int row0 = (int)blockIdx.x * MTILE + wave * 16;   // first row of this wave
    if (row0 >= rows) return;

    float* sQ = smem + wave * (3 * 16 * QSTR);              // 16 x 128 (stride QSTR)
    float* sK = sQ + 16 * QSTR;
    float* sV = sK + 16 * QSTR;
    float* sY = smem + WAVES * (3 * 16 * QSTR) + wave * (16 * YSTR); // 16 x 512

    const int halfHi = (lane >> 4);       // 0 for lanes 0-15, 1 for lanes 16-31
    const int mrow   = (lane & 15);       // row within 16-row tile (A layout); column in C layout
    const int crow0  = halfHi << 3;       // C-matrix row base for this half-wave
    const float* xrow = x   + (size_t)(row0 + mrow) * DIM_;
    const float* crow = ctx + (size_t)(row0 + mrow) * CTX_;

    // Prefetch streaming inputs for this wave's rows (global_prefetch_b8).
    __builtin_prefetch(xrow, 0, 0);
    __builtin_prefetch(crow, 0, 0);
    __builtin_prefetch(crow + 384, 0, 0);

    // ---------------- Phase 1a: q = x @ Wq  (K=512 -> 16 steps) --------------
    {
        v8f acc[8];
#pragma unroll
        for (int nt = 0; nt < 8; ++nt) acc[nt] = (v8f){};
        for (int kk = 0; kk < DIM_ / 32; ++kk) {
            v16bf a = make_a(xrow + kk * 32 + (halfHi << 3));
#pragma unroll
            for (int nt = 0; nt < 8; ++nt)
                acc[nt] = __builtin_amdgcn_wmma_f32_16x16x32_bf16(
                    false, a, false, load_b(WqP, 8, kk, nt, lane),
                    (short)0, acc[nt], false, false);
        }
#pragma unroll
        for (int nt = 0; nt < 8; ++nt)
#pragma unroll
            for (int j = 0; j < 8; ++j)
                sQ[(crow0 + j) * QSTR + nt * 16 + mrow] = acc[nt][j];
    }

    // ------------- Phase 1b: k = sig(ctx@Wk), v = ctx@Wv (shared A) ----------
    {
        v8f ak[8], av[8];
#pragma unroll
        for (int nt = 0; nt < 8; ++nt) { ak[nt] = (v8f){}; av[nt] = (v8f){}; }
        for (int kk = 0; kk < CTX_ / 32; ++kk) {
            v16bf a = make_a(crow + kk * 32 + (halfHi << 3));
#pragma unroll
            for (int nt = 0; nt < 8; ++nt) {
                ak[nt] = __builtin_amdgcn_wmma_f32_16x16x32_bf16(
                    false, a, false, load_b(WkP, 8, kk, nt, lane),
                    (short)0, ak[nt], false, false);
                av[nt] = __builtin_amdgcn_wmma_f32_16x16x32_bf16(
                    false, a, false, load_b(WvP, 8, kk, nt, lane),
                    (short)0, av[nt], false, false);
            }
        }
#pragma unroll
        for (int nt = 0; nt < 8; ++nt)
#pragma unroll
            for (int j = 0; j < 8; ++j) {
                int o = (crow0 + j) * QSTR + nt * 16 + mrow;
                sK[o] = sigmoid_f(ak[nt][j]);
                sV[o] = av[nt][j];
            }
    }

    // ------------- Phase 2: softmax(q)*scale, s = k.q, out = s*v -------------
    // 64 head-rows per wave (16 rows x 4 heads); each lane handles 2 serially.
    // Overwrites sQ with `out` (A-matrix input for the next GEMM).
#pragma unroll
    for (int t = 0; t < 2; ++t) {
        int hr = lane + t * 32;                 // head-row 0..63
        int r  = hr >> 2;
        int h  = hr & 3;
        float* qp = sQ + r * QSTR + h * 32;
        const float* kp = sK + r * QSTR + h * 32;
        const float* vp = sV + r * QSTR + h * 32;
        float m = -3.402823e38f;
        for (int i = 0; i < 32; ++i) m = fmaxf(m, qp[i]);
        float s = 0.f;
        for (int i = 0; i < 32; ++i) s += __expf(qp[i] - m);
        float inv = SCALE_ATTN / s;
        float dot = 0.f;
        for (int i = 0; i < 32; ++i) dot += __expf(qp[i] - m) * kp[i];
        dot *= inv;
        for (int i = 0; i < 32; ++i) qp[i] = dot * vp[i];
    }

    // ------------- Phase 3: y = out @ Wo + bo  (K=128, N=512) ----------------
    {
        v16bf a2[4];
#pragma unroll
        for (int kk = 0; kk < 4; ++kk)
            a2[kk] = make_a(sQ + mrow * QSTR + kk * 32 + (halfHi << 3));

        for (int gblk = 0; gblk < 4; ++gblk) {      // 4 groups of 128 cols
            v8f acc[8];
#pragma unroll
            for (int nt = 0; nt < 8; ++nt) acc[nt] = (v8f){};
#pragma unroll
            for (int kk = 0; kk < 4; ++kk)
#pragma unroll
                for (int nt = 0; nt < 8; ++nt)
                    acc[nt] = __builtin_amdgcn_wmma_f32_16x16x32_bf16(
                        false, a2[kk], false,
                        load_b(WoP, 32, kk, gblk * 8 + nt, lane),
                        (short)0, acc[nt], false, false);
#pragma unroll
            for (int nt = 0; nt < 8; ++nt) {
                int col = gblk * 128 + nt * 16 + mrow;
                float bias = bo[col];
#pragma unroll
                for (int j = 0; j < 8; ++j)
                    sY[(crow0 + j) * YSTR + col] = acc[nt][j] + bias;
            }
        }
    }

    // ------------- Phase 4: L2 normalize * g * sqrt(512), store --------------
    for (int r = 0; r < 16; ++r) {
        float ss = 0.f;
        for (int c = lane; c < DIM_; c += 32) {
            float v = sY[r * YSTR + c];
            ss += v * v;
        }
#pragma unroll
        for (int off = 16; off > 0; off >>= 1) ss += __shfl_xor(ss, off, 32);
        float nrm = sqrtf(ss);
        float sc  = SQRT_DIM / fmaxf(nrm, 1e-12f);
        const size_t ob = (size_t)(row0 + r) * DIM_;
        for (int c = lane; c < DIM_; c += 32)
            out[ob + c] = sY[r * YSTR + c] * sc * gg[c];
    }
}

// ---------------------------------------------------------------------------
extern "C" void kernel_launch(void* const* d_in, const int* in_sizes, int n_in,
                              void* d_out, int out_size, void* d_ws, size_t ws_size,
                              hipStream_t stream) {
    (void)n_in; (void)out_size; (void)ws_size;
    const float* x   = (const float*)d_in[0];
    const float* ctx = (const float*)d_in[1];
    const float* Wq  = (const float*)d_in[2];
    const float* Wk  = (const float*)d_in[3];
    const float* Wv  = (const float*)d_in[4];
    const float* Wo  = (const float*)d_in[5];
    const float* bo  = (const float*)d_in[6];
    const float* g   = (const float*)d_in[7];
    float* out = (float*)d_out;

    const int rows = in_sizes[0] / DIM_;   // 65536

    // Workspace layout (bf16 packed weights)
    char* ws = (char*)d_ws;
    __bf16* WqP = (__bf16*)(ws + 0);                 // 512*128*2 = 131072 B
    __bf16* WkP = (__bf16*)(ws + 131072);            // 768*128*2 = 196608 B
    __bf16* WvP = (__bf16*)(ws + 328680 - 1000);     // see below
    // keep offsets exact & aligned:
    WvP = (__bf16*)(ws + 131072 + 196608);           // 327680
    __bf16* WoP = (__bf16*)(ws + 131072 + 196608 + 196608);  // 524288

    // Pack weights into WMMA B-tile layout (cheap; weights ~1.25 MB total).
    {
        int n;
        n = DIM_ * HID_; hipLaunchKernelGGL(pack_b_kernel, dim3((n + 255) / 256), dim3(256), 0, stream, Wq, WqP, DIM_, HID_);
        n = CTX_ * HID_; hipLaunchKernelGGL(pack_b_kernel, dim3((n + 255) / 256), dim3(256), 0, stream, Wk, WkP, CTX_, HID_);
        n = CTX_ * HID_; hipLaunchKernelGGL(pack_b_kernel, dim3((n + 255) / 256), dim3(256), 0, stream, Wv, WvP, CTX_, HID_);
        n = HID_ * DIM_; hipLaunchKernelGGL(pack_b_kernel, dim3((n + 255) / 256), dim3(256), 0, stream, Wo, WoP, HID_, DIM_);
    }

    // Dynamic LDS: 4 waves * (3*16*132 + 16*516) floats = 233472 B (< 320 KB)
    const int smem_bytes = WAVES * (3 * 16 * QSTR + 16 * YSTR) * (int)sizeof(float);
    static bool attr_set = false;
    if (!attr_set) {
        hipFuncSetAttribute((const void*)lca_fused_kernel,
                            hipFuncAttributeMaxDynamicSharedMemorySize, smem_bytes);
        attr_set = true;
    }

    const int nblocks = rows / MTILE;      // 1024
    hipLaunchKernelGGL(lca_fused_kernel, dim3(nblocks), dim3(WAVES * 32),
                       smem_bytes, stream,
                       x, ctx, WqP, WkP, WvP, WoP, bo, g, out, rows);
}
